// GTV_84662395339188
// MI455X (gfx1250) — matrix operand order repaired and checked by reading, not compiled, not run on hardware
//
#include <hip/hip_runtime.h>
#include <cmath>

typedef __attribute__((ext_vector_type(16))) _Float16 v16h;
typedef __attribute__((ext_vector_type(8)))  float    v8f;

#define WTI  36
#define NPIX (WTI*WTI)   // 1296
#define NB   128
#define LROW 40          // LDS row pitch in halves (80B: bank-conflict-free-ish)

static inline int cdiv(int a, int b){ return (a + b - 1) / b; }

// ---------------------------------------------------------------------------
// Weight prep: f32 OIHW conv weights -> f16 TRANSPOSED im2col [Npad][KpadA]
// (k-major rows so both LDS staging and B-fragment reads are contiguous).
// k = c_in*9 + tap, row n = c_out; zero padded over real K/N.
// ---------------------------------------------------------------------------
__global__ void wconv_kernel(const float* __restrict__ w, _Float16* __restrict__ dst,
                             int Cin, int Cout, int Kpad, int KpadA, int Npad){
  int i = blockIdx.x*blockDim.x + threadIdx.x;
  if (i >= Npad*Kpad) return;
  int n = i / Kpad, k = i - n*Kpad;
  float v = 0.f;
  if (k < Cin*9 && n < Cout){
    int c = k/9, tap = k - c*9;
    v = w[((n*Cin)+c)*9 + tap];
  }
  dst[(long)n*KpadA + k] = (_Float16)v;
}

// f32 [K][Ncols] linear weight -> f16 transposed [Ncols][KpadA]
__global__ void wlin_kernel(const float* __restrict__ w, _Float16* __restrict__ dst,
                            int K, int Kpad, int KpadA, int Ncols){
  int i = blockIdx.x*blockDim.x + threadIdx.x;
  if (i >= Ncols*Kpad) return;
  int n = i / Kpad, k = i - n*Kpad;
  dst[(long)n*KpadA + k] = (_Float16)((k < K) ? w[(long)k*Ncols + n] : 0.f);
}

// f32 [M][K] activations -> f16 [M][Kpad], zero padded columns
__global__ void xf16_kernel(const float* __restrict__ x, _Float16* __restrict__ dst,
                            int K, int Kpad, int total){
  int i = blockIdx.x*blockDim.x + threadIdx.x;
  if (i >= total) return;
  int r = i / Kpad, k = i - r*Kpad;
  dst[i] = (_Float16)((k < K) ? x[(long)r*K + k] : 0.f);
}

// ---------------------------------------------------------------------------
// Implicit-GEMM 3x3 conv via V_WMMA_F32_16X16X32_F16.
// One wave computes a 16-spatial x (16*NT)-outchannel tile. Branchless
// clamp-based im2col staging; transposed weights -> all LDS traffic is b128.
// src pixel = (y + tapY + ofs, x + tapX + ofs); ofs=-1 same-pad, 0 valid.
// ---------------------------------------------------------------------------
template<int NT>
__global__ void __launch_bounds__(32)
conv3x3_wmma(const float* __restrict__ in, const _Float16* __restrict__ wfT,
             const float* __restrict__ bias, float* __restrict__ out,
             int Cin, int Hin, int Win,
             int Cout, int Hout, int Wout,
             int Kpad, int KpadA, int ofs, int doRelu,
             int outCtot, int ochOff)
{
  __shared__ _Float16 lA[16*LROW];
  __shared__ _Float16 lBt[16*NT*LROW];
  const int NTH = 16*NT;
  int lane  = threadIdx.x;
  int stile = blockIdx.x;
  int n0    = blockIdx.y * NTH;
  int b     = blockIdx.z;
  int S     = Hout*Wout;
  int Kreal = Cin*9;
  const float* inB = in + (long)b*Cin*Hin*Win;

  // staging role: lane owns A row m = lane>>1, K half kh = lane&1 (16 k's)
  int m  = lane >> 1;
  int kh = lane & 1;
  int p  = stile*16 + m;
  int pok = (p < S) ? 1 : 0;
  int pc = pok ? p : (S - 1);
  int y = pc / Wout, x = pc - y*Wout;

  // compute role
  int hf = lane >> 4;         // lane half
  int mr = lane & 15;         // A row / B col within tile

  v8f acc[NT];
  #pragma unroll
  for (int t = 0; t < NT; ++t) acc[t] = (v8f){};

  for (int k0 = 0; k0 < Kpad; k0 += 32){
    // ---- stage A [16 m][32 k]: branchless clamp + select ----
    int kbase = k0 + kh*16;
    int c   = kbase / 9;
    int tap = kbase - c*9;
    _Float16 tmp[16];
    #pragma unroll
    for (int i = 0; i < 16; ++i){
      int kok = (kbase + i < Kreal) ? 1 : 0;
      int dy = (tap >= 6) ? 2 : ((tap >= 3) ? 1 : 0);
      int dx = tap - 3*dy;
      int sy = y + dy + ofs, sx = x + dx + ofs;
      int ok = pok & kok & (sy >= 0) & (sy < Hin) & (sx >= 0) & (sx < Win);
      int cc  = kok ? c : (Cin - 1);
      int sy2 = min(max(sy, 0), Hin - 1);
      int sx2 = min(max(sx, 0), Win - 1);
      float v = inB[((long)cc*Hin + sy2)*Win + sx2];   // always-valid address
      tmp[i] = ok ? (_Float16)v : (_Float16)0.f;
      ++tap; if (tap == 9){ tap = 0; ++c; }
    }
    { // two aligned 16B LDS stores
      union { _Float16 h[16]; uint4 q[2]; } u;
      #pragma unroll
      for (int i = 0; i < 16; ++i) u.h[i] = tmp[i];
      uint4* dq = (uint4*)(lA + m*LROW + kh*16);
      dq[0] = u.q[0]; dq[1] = u.q[1];
    }
    // ---- stage B^T [NTH n-rows][32 k]: fully coalesced uint4 copies ----
    for (int r = lane; r < NTH; r += 32){
      const uint4* sq = (const uint4*)(wfT + (long)(n0 + r)*KpadA + k0);
      uint4* dq = (uint4*)(lBt + r*LROW);
      dq[0] = sq[0]; dq[1] = sq[1]; dq[2] = sq[2]; dq[3] = sq[3];
    }
    // prefetch next weight chunk (global_prefetch_b8); slack rows allocated
    __builtin_prefetch(wfT + (long)(n0 + lane)*KpadA + k0 + 32, 0, 1);
    __syncthreads();

    // ---- A fragment: two aligned 16B LDS reads ----
    union { v16h v; uint4 q[2]; } au;
    au.q[0] = *(const uint4*)(lA + mr*LROW + hf*16);
    au.q[1] = *(const uint4*)(lA + mr*LROW + hf*16 + 8);
    #pragma unroll
    for (int t = 0; t < NT; ++t){
      // ---- B fragment: two aligned 16B LDS reads (k-contiguous) ----
      union { v16h v; uint4 q[2]; } bu;
      const uint4* bq = (const uint4*)(lBt + (t*16 + mr)*LROW + hf*16);
      bu.q[0] = bq[0]; bu.q[1] = bq[1];
      acc[t] = __builtin_amdgcn_wmma_f32_16x16x32_f16(false, au.v, false, bu.v,
                                                      (short)0, acc[t], false, false);
    }
    __syncthreads();
  }

  // ---- store C: VGPR r -> M = r + 8*(lane>=16), N = lane&15 ----
  #pragma unroll
  for (int t = 0; t < NT; ++t){
    int nc = n0 + t*16 + mr;
    if (nc < Cout){
      float bv = bias[nc];
      float* outB = out + ((long)b*outCtot + ochOff + nc)*S;
      for (int r = 0; r < 8; ++r){
        int pp = stile*16 + hf*8 + r;
        if (pp < S){
          float v = acc[t][r] + bv;
          if (doRelu) v = fmaxf(v, 0.f);
          outB[pp] = v;
        }
      }
    }
  }
}

// ---------------------------------------------------------------------------
// Dense GEMM (M%16==0, N%(16*NT)==0, Kpad%32==0): out = relu(A*B + bias)
// A f16 [M][Kpad] (16B aligned rows); B f16 transposed [N][KpadA].
// ---------------------------------------------------------------------------
template<int NT>
__global__ void __launch_bounds__(32)
gemm_wmma(const _Float16* __restrict__ A, const _Float16* __restrict__ BwT,
          const float* __restrict__ bias, float* __restrict__ out,
          int Kpad, int KpadA, int Ncols, int doRelu)
{
  __shared__ _Float16 lBt[16*NT*LROW];
  const int NTH = 16*NT;
  int lane = threadIdx.x;
  int mtile = blockIdx.x;
  int n0 = blockIdx.y * NTH;
  int hf = lane >> 4, mr = lane & 15;
  const _Float16* Arow = A + (long)(mtile*16 + mr)*Kpad;
  v8f acc[NT];
  #pragma unroll
  for (int t = 0; t < NT; ++t) acc[t] = (v8f){};

  for (int k0 = 0; k0 < Kpad; k0 += 32){
    for (int r = lane; r < NTH; r += 32){   // stage B^T rows, coalesced
      const uint4* sq = (const uint4*)(BwT + (long)(n0 + r)*KpadA + k0);
      uint4* dq = (uint4*)(lBt + r*LROW);
      dq[0] = sq[0]; dq[1] = sq[1]; dq[2] = sq[2]; dq[3] = sq[3];
    }
    __builtin_prefetch(BwT + (long)(n0 + lane)*KpadA + k0 + 32, 0, 1);
    __syncthreads();

    union { v16h v; uint4 q[2]; } au;   // A frag: two aligned 16B global loads
    au.q[0] = *(const uint4*)(Arow + k0 + hf*16);
    au.q[1] = *(const uint4*)(Arow + k0 + hf*16 + 8);
    #pragma unroll
    for (int t = 0; t < NT; ++t){
      union { v16h v; uint4 q[2]; } bu;
      const uint4* bq = (const uint4*)(lBt + (t*16 + mr)*LROW + hf*16);
      bu.q[0] = bq[0]; bu.q[1] = bq[1];
      acc[t] = __builtin_amdgcn_wmma_f32_16x16x32_f16(false, au.v, false, bu.v,
                                                      (short)0, acc[t], false, false);
    }
    __syncthreads();
  }
  #pragma unroll
  for (int t = 0; t < NT; ++t){
    int nc = n0 + t*16 + mr;
    float bv = bias[nc];
    #pragma unroll
    for (int r = 0; r < 8; ++r){
      int mm = mtile*16 + hf*8 + r;
      float v = acc[t][r] + bv;
      if (doRelu) v = fmaxf(v, 0.f);
      out[(long)mm*Ncols + nc] = v;
    }
  }
}

// ---------------------------------------------------------------------------
// l2 head: lam[row] = clip(relu(h[row]·W2 + b2), 0.5, 100)
// ---------------------------------------------------------------------------
__global__ void lam_kernel(const float* __restrict__ h, const float* __restrict__ W2,
                           const float* __restrict__ b2, float* __restrict__ lam, int K){
  __shared__ float red[128];
  int row = blockIdx.x, t = threadIdx.x;
  float s = 0.f;
  for (int j = t; j < K; j += blockDim.x) s += h[(long)row*K + j]*W2[j];
  red[t] = s; __syncthreads();
  for (int o = 64; o > 0; o >>= 1){ if (t < o) red[t] += red[t+o]; __syncthreads(); }
  if (t == 0){
    float v = red[0] + b2[0];
    lam[row] = fminf(fmaxf(v, 0.5f), 100.f);
  }
}

// ---------------------------- elementwise helpers ---------------------------
__global__ void maxpool2_kernel(const float* __restrict__ in, float* __restrict__ out,
                                int n, int Hin, int Win){
  int i = blockIdx.x*blockDim.x + threadIdx.x; if (i >= n) return;
  int Wo = Win/2, S = (Hin/2)*Wo;
  int bc = i / S, r = i - bc*S, y = r / Wo, x = r - y*Wo;
  const float* p = in + (long)bc*Hin*Win + (2*y)*Win + 2*x;
  out[i] = fmaxf(fmaxf(p[0], p[1]), fmaxf(p[Win], p[Win+1]));
}

__global__ void up2_kernel(const float* __restrict__ in, float* __restrict__ out,
                           int n, int H, int W){
  int i = blockIdx.x*blockDim.x + threadIdx.x; if (i >= n) return;
  int Ho = 2*H+2, Wo = 2*W+2, S = Ho*Wo;
  int bc = i / S, r = i - bc*S, y = r / Wo, x = r - y*Wo;
  int yy = y - 1; if (yy < 0) yy = 1; else if (yy >= 2*H) yy = 2*H - 2;
  int xx = x - 1; if (xx < 0) xx = 1; else if (xx >= 2*W) xx = 2*W - 2;
  out[i] = in[(long)bc*H*W + (yy >> 1)*W + (xx >> 1)];
}

__global__ void copych_kernel(const float* __restrict__ src, float* __restrict__ dst,
                              int n, int Cs, int S, int Ctot, int off){
  int i = blockIdx.x*blockDim.x + threadIdx.x; if (i >= n) return;
  int b = i / (Cs*S), r = i - b*Cs*S, c = r / S, p = r - c*S;
  dst[((long)b*Ctot + off + c)*S + p] = src[i];
}

__global__ void add_kernel(const float* __restrict__ a, const float* __restrict__ b,
                           float* __restrict__ o, int n){
  int i = blockIdx.x*blockDim.x + threadIdx.x; if (i < n) o[i] = a[i] + b[i];
}
__global__ void copy_kernel(const float* __restrict__ a, float* __restrict__ o, int n){
  int i = blockIdx.x*blockDim.x + threadIdx.x; if (i < n) o[i] = a[i];
}
__global__ void zero_kernel(float* __restrict__ p, int n){
  int i = blockIdx.x*blockDim.x + threadIdx.x; if (i < n) p[i] = 0.f;
}

__global__ void edge_kernel(const float* __restrict__ E, const int* __restrict__ ei,
                            const int* __restrict__ ej, float* __restrict__ W, int nE){
  int i = blockIdx.x*blockDim.x + threadIdx.x;
  if (i >= NB*nE) return;
  int b = i / nE, e = i - b*nE;
  int a = ei[e], c = ej[e];
  const float* Eb = E + (long)b*3*NPIX;
  float d = 0.f;
  for (int ch = 0; ch < 3; ++ch){
    float df = Eb[ch*NPIX + a] - Eb[ch*NPIX + c];
    d += df*df;
  }
  atomicAdd(&W[(long)b*NPIX + a], expf(-0.5f*d));
}

__global__ void pgd_kernel(const float* __restrict__ v, const float* __restrict__ y,
                           const float* __restrict__ W, const float* __restrict__ lam,
                           float* __restrict__ out, int n){
  int i = blockIdx.x*blockDim.x + threadIdx.x; if (i >= n) return;
  int b = i / (3*NPIX), ch = (i / NPIX) % 3, p = i % NPIX;
  float eta = lam[b*3 + ch];
  float w   = W[(long)b*NPIX + p];
  float vv = v[i], yy = y[i];
  float z = vv - eta*(2.f*vv - 2.f*yy);
  float thr = fabsf(eta*w);
  float sg = (z > 0.f) ? 1.f : ((z < 0.f) ? -1.f : 0.f);
  float m1 = (fabsf(z) - thr > 0.f) ? 1.f : 0.f;
  z = z - m1*eta*w*sg;
  float m2 = (fabsf(z) - thr <= 0.f) ? 1.f : 0.f;
  out[i] = z - m2*z;
}

__global__ void mom_kernel(float* __restrict__ v, const float* __restrict__ o,
                           float c, int n){
  int i = blockIdx.x*blockDim.x + threadIdx.x; if (i < n) v[i] += c*(o[i] - v[i]);
}

// ---------------------------------------------------------------------------
extern "C" void kernel_launch(void* const* d_in, const int* in_sizes, int n_in,
                              void* d_out, int out_size, void* d_ws, size_t ws_size,
                              hipStream_t stream)
{
  (void)n_in; (void)out_size; (void)ws_size;
  const int Mrows = NB*3;

  // ---- input index maps: insertion-order vs jax sorted-key flattening ----
  int I_xf, I_ei, I_ej, IW[17], IB[17], IP[5][4];
  if (in_sizes[0] == NB*3*NPIX){                // xf first -> insertion order
    I_xf = 0; I_ei = 1; I_ej = 2;
    for (int j = 0; j < 17; ++j){ IW[j] = 3 + 2*j; IB[j] = IW[j] + 1; }
    for (int k = 0; k < 5; ++k) for (int q = 0; q < 4; ++q) IP[k][q] = 37 + 4*k + q;
  } else {                                      // sorted-key (jax tree) order
    I_ei = 0; I_ej = 1; I_xf = 56;
    static const int ws17[17] = {2,4,8,6,12,10,24,14,16,26,18,20,22,48,50,52,54};
    for (int j = 0; j < 17; ++j){ IW[j] = ws17[j]; IB[j] = ws17[j] + 1; }
    for (int k = 0; k < 5; ++k) for (int q = 0; q < 4; ++q) IP[k][q] = 28 + 4*k + q;
  }

  static const int cCin[17]  = {3,32,32,64,64,64,128,128,64,64,64,32,32,3,32,32,32};
  static const int cCout[17] = {32,32,64,64,64,128,64,64,64,32,32,32,3,32,32,32,3};

  char* wsb = (char*)d_ws; size_t off = 0;
  auto alloc = [&](size_t bytes)->void*{
    off = (off + 255) & ~(size_t)255;
    void* p = wsb + off; off += bytes; return p;
  };

  // transposed weights [Npad+32 rows][KpadA], +slack for branch-free prefetch
  _Float16* wf[17]; int Kp[17], KpA[17], Np[17];
  for (int j = 0; j < 17; ++j){
    Kp[j]  = ((cCin[j]*9 + 31)/32)*32;
    KpA[j] = Kp[j] + 32;
    Np[j]  = ((cCout[j] + 15)/16)*16;
    wf[j]  = (_Float16*)alloc((size_t)(Np[j] + 32)*KpA[j]*sizeof(_Float16));
  }
  const int KL = 1296, KLp = 1312, KLpA = KLp + 32, NL = 400;
  _Float16* wl[5];
  for (int k = 0; k < 5; ++k) wl[k] = (_Float16*)alloc((size_t)(NL + 32)*KLpA*sizeof(_Float16));

  float* T32a = (float*)alloc((size_t)NB*32*NPIX*4);
  float* H1   = (float*)alloc((size_t)NB*32*NPIX*4);
  float* BIG1 = (float*)alloc((size_t)NB*64*NPIX*4);   // t2, later h5 (64ch concat)
  float* P1   = (float*)alloc((size_t)NB*64*324*4);
  float* H2   = (float*)alloc((size_t)NB*64*324*4);
  float* P2   = (float*)alloc((size_t)NB*64*81*4);
  float* P3   = (float*)alloc((size_t)NB*128*81*4);
  float* U    = (float*)alloc((size_t)NB*128*400*4);   // up2(h3) 20x20
  float* H4   = (float*)alloc((size_t)NB*128*324*4);   // concat buffer (128ch)
  float* U2   = P2;  // aliases P2..H4 (13.85M floats) for up2(h4b) 38x38 (11.83M floats)
  float* Ebuf = (float*)alloc((size_t)NB*3*NPIX*4);
  float* Ybuf = (float*)alloc((size_t)NB*3*NPIX*4);
  float* Xbuf = (float*)alloc((size_t)NB*3*NPIX*4);
  float* Vbuf = (float*)alloc((size_t)NB*3*NPIX*4);
  float* Wb   = (float*)alloc((size_t)NB*NPIX*4);
  float* lam  = (float*)alloc((size_t)Mrows*4);
  float* H400 = (float*)alloc((size_t)Mrows*NL*4);
  _Float16* Af16 = (_Float16*)alloc((size_t)Mrows*KLp*sizeof(_Float16));
  float* OUT = (float*)d_out;

  const float* xf = (const float*)d_in[I_xf];

  for (int j = 0; j < 17; ++j){
    int tot = Np[j]*Kp[j];
    wconv_kernel<<<cdiv(tot,256),256,0,stream>>>((const float*)d_in[IW[j]], wf[j],
                                                 cCin[j], cCout[j], Kp[j], KpA[j], Np[j]);
  }
  for (int k = 0; k < 5; ++k)
    wlin_kernel<<<cdiv(NL*KLp,256),256,0,stream>>>((const float*)d_in[IP[k][0]], wl[k],
                                                   KL, KLp, KLpA, NL);

  auto conv = [&](const float* in, int j, float* o, int Hin, int Win, int Hout, int Wout,
                  int ofs, int doRelu, int outCtot, int ochOff){
    int S = Hout*Wout;
    int NT = (Np[j] >= 64) ? 4 : ((Np[j] == 32) ? 2 : 1);
    dim3 grid(cdiv(S,16), Np[j]/(16*NT), NB);
    switch (NT){
      case 4:
        conv3x3_wmma<4><<<grid,32,0,stream>>>(in, wf[j], (const float*)d_in[IB[j]], o,
            cCin[j], Hin, Win, cCout[j], Hout, Wout, Kp[j], KpA[j], ofs, doRelu,
            outCtot, ochOff);
        break;
      case 2:
        conv3x3_wmma<2><<<grid,32,0,stream>>>(in, wf[j], (const float*)d_in[IB[j]], o,
            cCin[j], Hin, Win, cCout[j], Hout, Wout, Kp[j], KpA[j], ofs, doRelu,
            outCtot, ochOff);
        break;
      default:
        conv3x3_wmma<1><<<grid,32,0,stream>>>(in, wf[j], (const float*)d_in[IB[j]], o,
            cCin[j], Hin, Win, cCout[j], Hout, Wout, Kp[j], KpA[j], ofs, doRelu,
            outCtot, ochOff);
        break;
    }
  };
  auto cnnp = [&](const float* src, int k){
    int tot = Mrows*KLp;
    xf16_kernel<<<cdiv(tot,256),256,0,stream>>>(src, Af16, KL, KLp, tot);
    dim3 g(Mrows/16, NL/(16*5));
    gemm_wmma<5><<<g,32,0,stream>>>(Af16, wl[k], (const float*)d_in[IP[k][1]], H400,
                                    KLp, KLpA, NL, 1);
    lam_kernel<<<Mrows,128,0,stream>>>(H400, (const float*)d_in[IP[k][2]],
                                       (const float*)d_in[IP[k][3]], lam, NL);
  };

  // ================= _cnnf =================
  conv(xf,   0, T32a, 36,36,36,36, -1, 1, 32, 0);                 // c1a
  conv(T32a, 1, H1,   36,36,36,36, -1, 1, 32, 0);                 // c1b -> h1
  conv(H1,   2, BIG1, 36,36,36,36, -1, 1, 64, 0);                 // c2a -> t2
  maxpool2_kernel<<<cdiv(NB*64*324,256),256,0,stream>>>(BIG1, P1, NB*64*324, 36, 36);
  conv(P1,   3, H2,   18,18,18,18, -1, 1, 64, 0);                 // c2  -> h2
  conv(H2,   4, P1,   18,18,18,18, -1, 1, 64, 0);                 // c3a -> t3
  maxpool2_kernel<<<cdiv(NB*64*81,256),256,0,stream>>>(P1, P2, NB*64*81, 18, 18);
  conv(P2,   5, P3,    9, 9, 9, 9, -1, 1,128, 0);                 // c3  -> h3
  up2_kernel<<<cdiv(NB*128*400,256),256,0,stream>>>(P3, U, NB*128*400, 9, 9);
  conv(U,    6, H4,   20,20,18,18,  0, 0,128, 0);                 // d1 (valid, no relu)
  copych_kernel<<<cdiv(NB*64*324,256),256,0,stream>>>(H2, H4, NB*64*324, 64, 324, 128, 64);
  conv(H4,   7, P1,   18,18,18,18, -1, 1, 64, 0);                 // c4a
  conv(P1,   8, H2,   18,18,18,18, -1, 1, 64, 0);                 // c4b -> h4b
  up2_kernel<<<cdiv(NB*64*1444,256),256,0,stream>>>(H2, U2, NB*64*1444, 18, 18);
  conv(U2,   9, BIG1, 38,38,36,36,  0, 0, 64, 0);                 // d2 -> h5[0:32]
  copych_kernel<<<cdiv(NB*32*NPIX,256),256,0,stream>>>(H1, BIG1, NB*32*NPIX, 32, NPIX, 64, 32);
  conv(BIG1,10, T32a, 36,36,36,36, -1, 1, 32, 0);                 // c5a
  conv(T32a,11, H1,   36,36,36,36, -1, 1, 32, 0);                 // c5b
  conv(H1,  12, Ebuf, 36,36,36,36, -1, 0,  3, 0);                 // c5c -> E

  // ================= _cnny =================
  conv(xf,  13, T32a, 36,36,36,36, -1, 1, 32, 0);                 // y1
  conv(T32a,14, H1,   36,36,36,36, -1, 1, 32, 0);                 // y2
  conv(H1,  15, T32a, 36,36,36,36, -1, 1, 32, 0);                 // y3
  conv(T32a,16, Ybuf, 36,36,36,36, -1, 0,  3, 0);                 // y4
  int tot3 = NB*3*NPIX;
  add_kernel<<<cdiv(tot3,256),256,0,stream>>>(xf, Ybuf, Xbuf, tot3);   // x = xf + cnn

  // ================= edge weights =================
  int nE = in_sizes[I_ei];
  zero_kernel<<<cdiv(NB*NPIX,256),256,0,stream>>>(Wb, NB*NPIX);
  edge_kernel<<<cdiv(NB*nE,256),256,0,stream>>>(Ebuf, (const int*)d_in[I_ei],
                                                (const int*)d_in[I_ej], Wb, nE);

  // ================= FISTA / PGD =================
  double t = 1.0; double cs[3];
  for (int i = 0; i < 3; ++i){
    double ot = t;
    t = (1.0 + sqrt(1.0 + 4.0*t*t)) / 2.0;
    cs[i] = (ot - 1.0) / t;
  }

  copy_kernel<<<cdiv(tot3,256),256,0,stream>>>(Xbuf, Vbuf, tot3);      // v = x
  cnnp(xf, 0);                                                        // lam = p1(xf)
  pgd_kernel<<<cdiv(tot3,256),256,0,stream>>>(Vbuf, Xbuf, Wb, lam, OUT, tot3);
  for (int i = 0; i < 3; ++i){                                        // p2, p3, p4
    mom_kernel<<<cdiv(tot3,256),256,0,stream>>>(Vbuf, OUT, (float)cs[i], tot3);
    cnnp(OUT, i+1);
    pgd_kernel<<<cdiv(tot3,256),256,0,stream>>>(Vbuf, Xbuf, Wb, lam, OUT, tot3);
  }
  cnnp(OUT, 4);                                                       // p5
  pgd_kernel<<<cdiv(tot3,256),256,0,stream>>>(Vbuf, Xbuf, Wb, lam, OUT, tot3);
}
